// ATTNEncoder_73641509257250
// MI455X (gfx1250) — compile-verified, matrix operand order checked by reading, not run on hardware
//
#include <hip/hip_runtime.h>

// ---------------------------------------------------------------------------
// CDNA5 / gfx1250 fused transformer-encoder pipeline, round 2.
// bf16 operands end-to-end for GEMMs (converted once by producers),
// V_WMMA_F32_16X16X32_BF16 with f32 accumulate, LDS double-buffered
// 128x128x32 tiles, wave32, 8 waves/block.
// ---------------------------------------------------------------------------

typedef __attribute__((ext_vector_type(16))) __bf16 v16bf;
typedef __attribute__((ext_vector_type(8)))  float  v8f;

#define Bc   32
#define Sc   512
#define Ec   512
#define DKc  128
#define Hc   6
#define D2c  1024

#define BM  128
#define BN  128
#define BK  32
#define BKP 40   // padded bf16 row stride in LDS (80B: conflict-free, 16B aligned)

typedef unsigned short bfu;   // raw bf16 bits

__device__ __forceinline__ bfu f2bf(float f) {
  union { float f; unsigned u; } x; x.f = f;
  unsigned r = x.u + 0x7FFFu + ((x.u >> 16) & 1u);   // round-to-nearest-even
  return (bfu)(r >> 16);
}

// ---------------------------------------------------------------------------
// Generic batched GEMM on bf16 operands:
//   Cf / Cbf = act( alpha * A[z] (MxK) * B[z] (KxN) + bias )
// Two-level batch z = b1*nb2+b2 with independent strides (heads, batch).
// transB=1: B stored [N][K] (e.g. K-matrix for Q*K^T). LDS double-buffered.
// ---------------------------------------------------------------------------
__global__ __launch_bounds__(256)
void gemm_bf16_wmma(const bfu* __restrict__ A, int lda, long long sA1, long long sA2,
                    const bfu* __restrict__ Bm, int ldb, int transB, long long sB1, long long sB2,
                    const float* __restrict__ bias, long long sBias1,
                    float* __restrict__ Cf, bfu* __restrict__ Cbf,
                    int ldc, long long sC1, long long sC2,
                    int M, int N, int Kd, int nb2, float alpha, int actTanh)
{
  __shared__ __align__(16) bfu sA[2 * BM * BKP];
  __shared__ __align__(16) bfu sB[2 * BN * BKP];

  const int z  = blockIdx.z;
  const int b1 = z / nb2;
  const int b2 = z % nb2;
  A  += b1 * sA1 + b2 * sA2;
  Bm += b1 * sB1 + b2 * sB2;
  if (Cf)  Cf  += b1 * sC1 + b2 * sC2;
  if (Cbf) Cbf += b1 * sC1 + b2 * sC2;
  const float* biasP = bias ? (bias + b1 * sBias1) : nullptr;

  const int rowBase = blockIdx.y * BM;
  const int colBase = blockIdx.x * BN;
  const int tid   = threadIdx.x;
  const int wave  = tid >> 5;
  const int lane  = tid & 31;
  const int wm    = wave >> 1;       // 4 row groups of 32
  const int wn    = wave & 1;        // 2 col groups of 64
  const int lrow  = lane & 15;
  const int lhalf = lane >> 4;

  v8f acc[2][4];
  for (int i = 0; i < 2; ++i)
    for (int j = 0; j < 4; ++j)
      for (int r = 0; r < 8; ++r) acc[i][j][r] = 0.0f;

  // ---- tile staging: pure bf16 copies (no conversion in the hot loop) ----
  auto stage = [&](int kk, int buf) {
    bfu* sAb = sA + buf * (BM * BKP);
    bfu* sBb = sB + buf * (BN * BKP);
    #pragma unroll
    for (int it = 0; it < 2; ++it) {               // A: BM x BK, 8 bf16/thread
      int lin = (it * 256 + tid) * 8;
      int r = lin >> 5;
      int c = lin & (BK - 1);
      uint4 v = *reinterpret_cast<const uint4*>(A + (long long)(rowBase + r) * lda + kk + c);
      *reinterpret_cast<uint4*>(&sAb[r * BKP + c]) = v;
    }
    if (transB) {                                  // B is [N][K]: straight copy
      #pragma unroll
      for (int it = 0; it < 2; ++it) {
        int lin = (it * 256 + tid) * 8;
        int n = lin >> 5;
        int c = lin & (BK - 1);
        uint4 v = *reinterpret_cast<const uint4*>(Bm + (long long)(colBase + n) * ldb + kk + c);
        *reinterpret_cast<uint4*>(&sBb[n * BKP + c]) = v;
      }
    } else {                                       // B is [K][N]: transpose to [n][k]
      #pragma unroll
      for (int it = 0; it < 2; ++it) {
        int lin = (it * 256 + tid) * 8;
        int k = lin >> 7;
        int n = lin & (BN - 1);
        union { uint4 q; bfu u[8]; } v;
        v.q = *reinterpret_cast<const uint4*>(Bm + (long long)(kk + k) * ldb + colBase + n);
        #pragma unroll
        for (int j = 0; j < 8; ++j) sBb[(n + j) * BKP + k] = v.u[j];
      }
    }
  };

  stage(0, 0);
  int cur = 0;
  for (int k0 = 0; k0 < Kd; k0 += BK) {
    __syncthreads();                               // staged tile visible

    // ---- fragment loads (two ds_load_b128 per fragment) ------------------
    // 16-bit A layout (ISA 7.12.2): lane = row M, half-wave selects K group:
    // dwords0-3 <- K = half*8..+7, dwords4-7 <- K = 16+half*8..+7.
    const bfu* sAb = sA + cur * (BM * BKP);
    const bfu* sBb = sB + cur * (BN * BKP);
    union Frag { uint4 q[2]; v16bf v; };
    Frag af[2], bfr[4];
    #pragma unroll
    for (int i = 0; i < 2; ++i) {
      int m = wm * 32 + i * 16 + lrow;
      af[i].q[0] = *reinterpret_cast<const uint4*>(&sAb[m * BKP + lhalf * 8]);
      af[i].q[1] = *reinterpret_cast<const uint4*>(&sAb[m * BKP + 16 + lhalf * 8]);
    }
    #pragma unroll
    for (int j = 0; j < 4; ++j) {
      int n = wn * 64 + j * 16 + lrow;
      bfr[j].q[0] = *reinterpret_cast<const uint4*>(&sBb[n * BKP + lhalf * 8]);
      bfr[j].q[1] = *reinterpret_cast<const uint4*>(&sBb[n * BKP + 16 + lhalf * 8]);
    }

    // prefetch next tile into the other LDS buffer while WMMAs run
    if (k0 + BK < Kd) stage(k0 + BK, cur ^ 1);

    #pragma unroll
    for (int i = 0; i < 2; ++i)
      #pragma unroll
      for (int j = 0; j < 4; ++j)
        acc[i][j] = __builtin_amdgcn_wmma_f32_16x16x32_bf16(
            false, af[i].v, false, bfr[j].v, (short)0, acc[i][j], false, false);
    cur ^= 1;
  }

  // ---- epilogue: alpha, bias, tanh; dual f32/bf16 stores -----------------
  // C/D layout: VGPR r -> M = r + 8*half, lane -> N = lane&15.
  #pragma unroll
  for (int i = 0; i < 2; ++i) {
    #pragma unroll
    for (int j = 0; j < 4; ++j) {
      int gm0 = rowBase + wm * 32 + i * 16 + 8 * lhalf;
      int gn  = colBase + wn * 64 + j * 16 + lrow;
      float bv = biasP ? biasP[gn] : 0.0f;
      #pragma unroll
      for (int r = 0; r < 8; ++r) {
        float x = acc[i][j][r] * alpha + bv;
        if (actTanh) x = tanhf(x);
        long long idx = (long long)(gm0 + r) * ldc + gn;
        if (Cf)  Cf[idx]  = x;
        if (Cbf) Cbf[idx] = f2bf(x);
      }
    }
  }
}

// ---------------------------------------------------------------------------
// one-time f32 -> bf16 conversion (weights); 8 elements/thread, b128 I/O
// ---------------------------------------------------------------------------
__global__ __launch_bounds__(256)
void cvt_bf16_kernel(const float* __restrict__ in, bfu* __restrict__ out)
{
  long long i = ((long long)blockIdx.x * 256 + threadIdx.x) * 8;
  float4 a = *reinterpret_cast<const float4*>(in + i);
  float4 b = *reinterpret_cast<const float4*>(in + i + 4);
  union { uint4 q; bfu u[8]; } o;
  o.u[0] = f2bf(a.x); o.u[1] = f2bf(a.y); o.u[2] = f2bf(a.z); o.u[3] = f2bf(a.w);
  o.u[4] = f2bf(b.x); o.u[5] = f2bf(b.y); o.u[6] = f2bf(b.z); o.u[7] = f2bf(b.w);
  *reinterpret_cast<uint4*>(out + i) = o.q;
}

// ---------------------------------------------------------------------------
// X + positional encoding; emits f32 (residual path) and bf16 (GEMM A)
// ---------------------------------------------------------------------------
__global__ __launch_bounds__(256)
void pos_add_kernel(const float* __restrict__ X, float* __restrict__ XP,
                    bfu* __restrict__ XPh)
{
  long long idx = (long long)blockIdx.x * 256 + threadIdx.x;
  int e = (int)(idx % Ec);
  int s = (int)((idx / Ec) % Sc);
  int c = e >> 1;
  float invf = (c < 128) ? 1.0f : 1e-4f;   // 10000^((2j)//d) in {1, 10000}
  float ang  = (float)s * invf;
  float pe   = (e & 1) ? cosf(ang) : sinf(ang);
  float v = X[idx] + pe;
  XP[idx]  = v;
  XPh[idx] = f2bf(v);
}

// ---------------------------------------------------------------------------
// softmax over the QUERY axis (axis=2 of (H,B,S,T)): normalize each column t.
// Reads/holds exp() in the f32 scores buffer, emits bf16 attn for the AV GEMM.
// ---------------------------------------------------------------------------
__global__ __launch_bounds__(256)
void softmax_query_kernel(float* __restrict__ sc, bfu* __restrict__ at)
{
  long long base = (long long)blockIdx.y * Sc * Sc;
  float* p = sc + base;
  bfu*   d = at + base;
  int t = blockIdx.x * 256 + threadIdx.x;
  float mx = -1e30f;
  for (int s = 0; s < Sc; ++s) mx = fmaxf(mx, p[(long long)s * Sc + t]);
  float sum = 0.0f;
  for (int s = 0; s < Sc; ++s) {
    float e = __expf(p[(long long)s * Sc + t] - mx);
    sum += e;
    p[(long long)s * Sc + t] = e;
  }
  float inv = 1.0f / sum;
  for (int s = 0; s < Sc; ++s) d[(long long)s * Sc + t] = f2bf(p[(long long)s * Sc + t] * inv);
}

// ---------------------------------------------------------------------------
// BatchNorm over axis=0 (batch) fused with residual add; dual f32/bf16 out.
// in2 has width w2 and wraps (handles concat([xp,xp]) when w2 = E).
// ---------------------------------------------------------------------------
__global__ __launch_bounds__(256)
void bn_add_kernel(const float* __restrict__ in1, const float* __restrict__ in2,
                   int w2, float* __restrict__ out, bfu* __restrict__ outh)
{
  long long idx = (long long)blockIdx.x * 256 + threadIdx.x;  // over S*D2
  int s = (int)(idx / D2c);
  int e = (int)(idx % D2c);
  long long i2 = (long long)s * w2 + (e % w2);
  const long long sd = (long long)Sc * D2c;
  const long long s2 = (long long)Sc * w2;
  float sum = 0.0f, sq = 0.0f;
  for (int b = 0; b < Bc; ++b) {
    float v = in1[b * sd + idx] + in2[b * s2 + i2];
    sum += v; sq += v * v;
  }
  float mean = sum * (1.0f / Bc);
  float var  = sq  * (1.0f / Bc) - mean * mean;
  float inv  = rsqrtf(var + 1e-3f);
  for (int b = 0; b < Bc; ++b) {
    float v = (in1[b * sd + idx] + in2[b * s2 + i2] - mean) * inv;
    out[b * sd + idx] = v;
    if (outh) outh[b * sd + idx] = f2bf(v);
  }
}

// ---------------------------------------------------------------------------
extern "C" void kernel_launch(void* const* d_in, const int* in_sizes, int n_in,
                              void* d_out, int out_size, void* d_ws, size_t ws_size,
                              hipStream_t stream)
{
  const float* X    = (const float*)d_in[0];
  const float* W1[3] = { (const float*)d_in[1], (const float*)d_in[2], (const float*)d_in[3] };
  const float* b1[3] = { (const float*)d_in[4], (const float*)d_in[5], (const float*)d_in[6] };
  const float* Wo1  = (const float*)d_in[7];
  const float* bo1  = (const float*)d_in[8];
  const float* Wff1 = (const float*)d_in[9];
  const float* bff1 = (const float*)d_in[10];
  const float* W2[3] = { (const float*)d_in[11], (const float*)d_in[12], (const float*)d_in[13] };
  const float* b2[3] = { (const float*)d_in[14], (const float*)d_in[15], (const float*)d_in[16] };
  const float* Wo2  = (const float*)d_in[17];
  const float* bo2  = (const float*)d_in[18];
  const float* Wff2 = (const float*)d_in[19];
  const float* bff2 = (const float*)d_in[20];
  float* OUT = (float*)d_out;

  const int M   = Bc * Sc;          // 16384 token rows
  const int HDK = Hc * DKc;         // 768

  // ---- workspace layout ---------------------------------------------------
  size_t off = 0;
  auto alloc = [&](size_t bytes) {
    void* p = (char*)d_ws + off;
    off += (bytes + 255) & ~(size_t)255;
    return p;
  };
  // f32
  float* XP = (float*)alloc((size_t)Bc * Sc * Ec * 4);          // 32 MB
  float* SC = (float*)alloc((size_t)Hc * Bc * Sc * Sc * 4);     // 192 MB (H,B,S,T)
  float* T  = (float*)alloc((size_t)Bc * Sc * D2c * 4);         // 64 MB
  float* N1 = (float*)alloc((size_t)Bc * Sc * D2c * 4);
  float* N2 = (float*)alloc((size_t)Bc * Sc * D2c * 4);
  float* N3 = (float*)alloc((size_t)Bc * Sc * D2c * 4);
  // bf16 activations
  bfu* XPh = (bfu*)alloc((size_t)Bc * Sc * Ec * 2);
  bfu* Qh  = (bfu*)alloc((size_t)Hc * Bc * Sc * DKc * 2);       // (H,B,S,DK)
  bfu* Kh  = (bfu*)alloc((size_t)Hc * Bc * Sc * DKc * 2);
  bfu* Vh  = (bfu*)alloc((size_t)Hc * Bc * Sc * DKc * 2);
  bfu* ATh = (bfu*)alloc((size_t)Hc * Bc * Sc * Sc * 2);        // 96 MB
  bfu* Oh  = (bfu*)alloc((size_t)Bc * Sc * HDK * 2);
  bfu* N1h = (bfu*)alloc((size_t)Bc * Sc * D2c * 2);
  bfu* N2h = (bfu*)alloc((size_t)Bc * Sc * D2c * 2);
  bfu* N3h = (bfu*)alloc((size_t)Bc * Sc * D2c * 2);
  // bf16 weights (converted once per launch)
  bfu* W1h[3], * W2h[3];
  for (int i = 0; i < 3; ++i) W1h[i] = (bfu*)alloc((size_t)Hc * Ec * DKc * 2);
  bfu* Wo1h  = (bfu*)alloc((size_t)HDK * D2c * 2);
  bfu* Wff1h = (bfu*)alloc((size_t)D2c * D2c * 2);
  for (int i = 0; i < 3; ++i) W2h[i] = (bfu*)alloc((size_t)Hc * D2c * DKc * 2);
  bfu* Wo2h  = (bfu*)alloc((size_t)HDK * D2c * 2);
  bfu* Wff2h = (bfu*)alloc((size_t)D2c * D2c * 2);
  (void)ws_size; (void)in_sizes; (void)n_in; (void)out_size;

  auto cvt = [&](const float* src, bfu* dst, size_t n) {
    cvt_bf16_kernel<<<(unsigned)(n / 2048), 256, 0, stream>>>(src, dst);
  };
  auto gemm = [&](const bfu* Ap, int lda, long long a1, long long a2,
                  const bfu* Bp, int ldb, int tB, long long bs1, long long bs2,
                  const float* biasp, long long biasS,
                  float* Cfp, bfu* Cbp, int ldc, long long c1, long long c2,
                  int Mm, int Nn, int Kd, int nb1, int nb2, float alpha, int act) {
    dim3 grid(Nn / BN, Mm / BM, nb1 * nb2);
    gemm_bf16_wmma<<<grid, 256, 0, stream>>>(Ap, lda, a1, a2, Bp, ldb, tB, bs1, bs2,
                                             biasp, biasS, Cfp, Cbp, ldc, c1, c2,
                                             Mm, Nn, Kd, nb2, alpha, act);
  };

  const long long sQh = (long long)Bc * Sc * DKc;   // head stride (H,B,S,DK)
  const long long sQb = (long long)Sc * DKc;
  const long long sSh = (long long)Bc * Sc * Sc;    // head stride (H,B,S,T)
  const long long sSb = (long long)Sc * Sc;
  const float scScale = 0.08838834764831845f;       // 1/sqrt(128)

  // ---- one-time conversions ----------------------------------------------
  for (int i = 0; i < 3; ++i) cvt(W1[i], W1h[i], (size_t)Hc * Ec * DKc);
  cvt(Wo1, Wo1h, (size_t)HDK * D2c);
  cvt(Wff1, Wff1h, (size_t)D2c * D2c);
  for (int i = 0; i < 3; ++i) cvt(W2[i], W2h[i], (size_t)Hc * D2c * DKc);
  cvt(Wo2, Wo2h, (size_t)HDK * D2c);
  cvt(Wff2, Wff2h, (size_t)D2c * D2c);
  pos_add_kernel<<<(Bc * Sc * Ec) / 256, 256, 0, stream>>>(X, XP, XPh);

  // ======================= layer 1 ========================================
  bfu* QKV[3] = { Qh, Kh, Vh };
  for (int i = 0; i < 3; ++i)
    gemm(XPh, Ec, 0, 0, W1h[i], DKc, 0, (long long)Ec * DKc, 0, b1[i], DKc,
         nullptr, QKV[i], DKc, (long long)M * DKc, 0, M, DKc, Ec, Hc, 1, 1.0f, 0);
  gemm(Qh, DKc, sQh, sQb, Kh, DKc, 1, sQh, sQb, nullptr, 0,
       SC, nullptr, Sc, sSh, sSb, Sc, Sc, DKc, Hc, Bc, scScale, 0);
  softmax_query_kernel<<<dim3(Sc / 256, Hc * Bc), 256, 0, stream>>>(SC, ATh);
  gemm(ATh, Sc, sSh, sSb, Vh, DKc, 0, sQh, sQb, nullptr, 0,
       nullptr, Oh, HDK, (long long)DKc, (long long)Sc * HDK,
       Sc, DKc, Sc, Hc, Bc, 1.0f, 0);
  gemm(Oh, HDK, 0, 0, Wo1h, D2c, 0, 0, 0, bo1, 0,
       T, nullptr, D2c, 0, 0, M, D2c, HDK, 1, 1, 1.0f, 0);
  bn_add_kernel<<<(Sc * D2c) / 256, 256, 0, stream>>>(T, XP, Ec, N1, N1h);
  gemm(N1h, D2c, 0, 0, Wff1h, D2c, 0, 0, 0, bff1, 0,
       T, nullptr, D2c, 0, 0, M, D2c, D2c, 1, 1, 1.0f, 1);
  bn_add_kernel<<<(Sc * D2c) / 256, 256, 0, stream>>>(T, N1, D2c, N2, N2h);

  // ======================= layer 2 ========================================
  const float* b2p[3] = { b2[0], b2[1], b2[2] };
  for (int i = 0; i < 3; ++i)
    gemm(N2h, D2c, 0, 0, W2h[i], DKc, 0, (long long)D2c * DKc, 0, b2p[i], DKc,
         nullptr, QKV[i], DKc, (long long)M * DKc, 0, M, DKc, D2c, Hc, 1, 1.0f, 0);
  gemm(Qh, DKc, sQh, sQb, Kh, DKc, 1, sQh, sQb, nullptr, 0,
       SC, nullptr, Sc, sSh, sSb, Sc, Sc, DKc, Hc, Bc, scScale, 0);
  softmax_query_kernel<<<dim3(Sc / 256, Hc * Bc), 256, 0, stream>>>(SC, ATh);
  gemm(ATh, Sc, sSh, sSb, Vh, DKc, 0, sQh, sQb, nullptr, 0,
       nullptr, Oh, HDK, (long long)DKc, (long long)Sc * HDK,
       Sc, DKc, Sc, Hc, Bc, 1.0f, 0);
  gemm(Oh, HDK, 0, 0, Wo2h, D2c, 0, 0, 0, bo2, 0,
       T, nullptr, D2c, 0, 0, M, D2c, HDK, 1, 1, 1.0f, 0);
  bn_add_kernel<<<(Sc * D2c) / 256, 256, 0, stream>>>(T, N2, D2c, N3, N3h);
  gemm(N3h, D2c, 0, 0, Wff2h, D2c, 0, 0, 0, bff2, 0,
       T, nullptr, D2c, 0, 0, M, D2c, D2c, 1, 1, 1.0f, 1);
  bn_add_kernel<<<(Sc * D2c) / 256, 256, 0, stream>>>(T, N3, D2c, OUT, nullptr);
}
